// BiAttentionLayer_25494925869170
// MI455X (gfx1250) — matrix operand order repaired and checked by reading, not compiled, not run on hardware
//
#include <hip/hip_runtime.h>

typedef __attribute__((ext_vector_type(16))) _Float16 v16h;
typedef __attribute__((ext_vector_type(8)))  _Float16 v8h;
typedef __attribute__((ext_vector_type(8)))  float    v8f;
typedef __attribute__((ext_vector_type(4)))  unsigned int v4u;
typedef __attribute__((ext_vector_type(8)))  int  v8i;
typedef __attribute__((ext_vector_type(4)))  int  v4i;

constexpr int D      = 100;     // feature dim
constexpr int NC     = 16384;   // context rows
constexpr int NQ     = 4096;    // question rows
constexpr int DPAD   = 128;     // K padded for WMMA (4 x 32)
constexpr int NTILES = 7;       // 7*16 = 112 >= 100 output-dim tiles
constexpr int CHUNK  = 64;      // question columns per inner step
constexpr int NCH    = NQ / CHUNK;   // 64 chunks
constexpr int RPB    = 128;     // context rows per block (8 waves x 16)
constexpr int WAVES  = 8;

// workspace layout (float offsets)
constexpr int WS_U    = 0;                  // U_A, [NC][D]
constexpr int WS_M    = WS_U + NC * D;      // per-row max of S, [NC]
constexpr int WS_CW1  = WS_M + NC;          // context @ w1, [NC]
constexpr int WS_QW2  = WS_CW1 + NC;        // question @ w2, [NQ]
constexpr int WS_MZ   = WS_QW2 + NQ;        // {max, sumexp} of m
constexpr int WS_H    = WS_MZ + 8;          // h = b @ context, [D] (reserve 128)
constexpr int WS_QBG  = WS_H + 128;         // f16 [NQ][DPAD], Q padded row-major
constexpr int WS_QTG  = WS_QBG + NQ * DPAD / 2; // f16 [DPAD][NQ], Q transposed

// ---------------- kernel 0a: bias dots cw1[i] = c_i.w1, qw2[j] = q_j.w2 -----
__global__ void k_bias(const float* __restrict__ ctx, const float* __restrict__ qst,
                       const float* __restrict__ kern, float* __restrict__ ws) {
  int idx = blockIdx.x * blockDim.x + threadIdx.x;
  if (idx < NC) {
    const float* r = ctx + idx * D;
    float s = 0.f;
    #pragma unroll 4
    for (int k = 0; k < D; ++k) s += r[k] * kern[k];
    ws[WS_CW1 + idx] = s;
  } else if (idx < NC + NQ) {
    int j = idx - NC;
    const float* r = qst + j * D;
    float s = 0.f;
    #pragma unroll 4
    for (int k = 0; k < D; ++k) s += r[k] * kern[D + k];
    ws[WS_QW2 + j] = s;
  }
}

// ---------------- kernel 0b: pre-convert Q to f16 (padded + transposed) -----
__global__ void k_prep(const float* __restrict__ qst, float* __restrict__ ws) {
  int idx = blockIdx.x * blockDim.x + threadIdx.x;
  if (idx >= NQ * DPAD) return;
  int j = idx >> 7, d = idx & 127;
  float v = (d < D) ? qst[j * D + d] : 0.f;
  _Float16 h = (_Float16)v;
  _Float16* qbg = (_Float16*)(ws + WS_QBG);
  _Float16* qtg = (_Float16*)(ws + WS_QTG);
  qbg[j * DPAD + d] = h;
  qtg[d * NQ + j]   = h;
}

// ---------------- TDM helper: issue one tensor_load_to_lds ------------------
// Descriptor per CDNA5 ISA (groups 0/1; groups 2/3 zero
// since tile_dim2 == 0). data_size = 3 (8-byte units). count=1, type=2.
__device__ __forceinline__ void tdm_load(unsigned lds_byte, unsigned long long gaddr,
                                         unsigned td0, unsigned td1,
                                         unsigned tile0, unsigned tile1,
                                         unsigned long long stride0) {
  v4u g0; v8i g1; v4i gz;
  gz[0] = 0; gz[1] = 0; gz[2] = 0; gz[3] = 0;
  g0[0] = 1u;                                       // count=1 (valid user D#)
  g0[1] = lds_byte;                                 // lds_addr
  g0[2] = (unsigned)gaddr;                          // global_addr[31:0]
  g0[3] = (unsigned)(gaddr >> 32) | 0x80000000u;    // global_addr[56:32] | type=2
  g1[0] = (int)(3u << 16);                          // data_size=8B, wg_mask=0
  g1[1] = (int)((td0 & 0xffffu) << 16);             // tensor_dim0[15:0]
  g1[2] = (int)((td0 >> 16) | ((td1 & 0xffffu) << 16)); // td0 hi | td1 lo
  g1[3] = (int)((td1 >> 16) | (tile0 << 16));       // td1 hi | tile_dim0
  g1[4] = (int)tile1;                               // tile_dim1 (tile_dim2=0)
  g1[5] = (int)(stride0 & 0xffffffffu);             // tensor_dim0_stride lo
  g1[6] = (int)((stride0 >> 32) & 0xffffu);         // stride hi | dim1_stride lo=0
  g1[7] = 0;
#if __has_include(<hip/amd_detail/amd_gfx1250_TDM.h>)
  v8i gz8; gz8[0]=0; gz8[1]=0; gz8[2]=0; gz8[3]=0; gz8[4]=0; gz8[5]=0; gz8[6]=0; gz8[7]=0;
  __builtin_amdgcn_tensor_load_to_lds(g0, g1, gz, gz, gz8, 0);
#else
  __builtin_amdgcn_tensor_load_to_lds(g0, g1, gz, gz, 0);
#endif
}

__device__ __forceinline__ unsigned lds_off(const void* p) {
  // generic LDS address: low 32 bits are the in-LDS byte offset
  return (unsigned)(unsigned long long)(uintptr_t)p;
}

// ---------------- kernel 1: flash-attention pass (WMMA + TDM) ---------------
__launch_bounds__(256)
__global__ void k_attn(const float* __restrict__ ctx, const float* __restrict__ kern,
                       float* __restrict__ ws) {
  __shared__ __attribute__((aligned(64))) _Float16 Qb[2][CHUNK * DPAD]; // [col][d]
  __shared__ __attribute__((aligned(64))) _Float16 Qt[2][DPAD * CHUNK]; // [d][col]
  __shared__ __attribute__((aligned(32))) _Float16 Plds[WAVES][16 * CHUNK];

  const int tid  = threadIdx.x;
  const int wave = tid >> 5;
  const int lane = tid & 31;
  const int lt   = lane & 15;   // 0..15
  const int lh   = lane >> 4;   // lane half
  const int rowBase = blockIdx.x * RPB + wave * 16;

  const _Float16* qbg = (const _Float16*)(ws + WS_QBG);
  const _Float16* qtg = (const _Float16*)(ws + WS_QTG);

  // A-fragments: rows of context*w3, K = 0..127 (zero padded), kept in VGPRs.
  v16h afrag[4];
  #pragma unroll
  for (int kk = 0; kk < 4; ++kk) {
    #pragma unroll
    for (int e = 0; e < 16; ++e) {
      int K = kk * 32 + lh * 16 + e;   // contiguous k order: k = 16*lh + e
      float v = (K < D) ? ctx[(rowBase + lt) * D + K] * kern[2 * D + K] : 0.f;
      afrag[kk][e] = (_Float16)v;
    }
  }
  float cw1v[8];
  #pragma unroll
  for (int g = 0; g < 8; ++g) cw1v[g] = ws[WS_CW1 + rowBase + lh * 8 + g];

  float mrun[8], lrun[8];
  v8f U[NTILES];
  v8f zero = {};
  #pragma unroll
  for (int g = 0; g < 8; ++g) { mrun[g] = -3.0e38f; lrun[g] = 0.f; }
  #pragma unroll
  for (int t = 0; t < NTILES; ++t) U[t] = zero;

  const unsigned qbL[2] = { lds_off(&Qb[0][0]), lds_off(&Qb[1][0]) };
  const unsigned qtL[2] = { lds_off(&Qt[0][0]), lds_off(&Qt[1][0]) };

  // issue TDM for chunk 0 (wave 0 drives the DMA; EXEC irrelevant for TDM)
  if (wave == 0) {
    // Qb tile: 1-D contiguous copy of CHUNK*DPAD f16 = 2048 x 8B
    tdm_load(qbL[0], (unsigned long long)(uintptr_t)(qbg),
             CHUNK * DPAD * 2 / 8, 0u, CHUNK * DPAD * 2 / 8, 0u, 0ull);
    // Qt tile: 2-D, 128 rows of 64 f16 (16 x 8B), row stride NQ f16 (1024 x 8B)
    tdm_load(qtL[0], (unsigned long long)(uintptr_t)(qtg),
             NQ * 2 / 8, DPAD, CHUNK * 2 / 8, DPAD, NQ * 2 / 8);
  }

  for (int c = 0; c < NCH; ++c) {
    const int j0 = c * CHUNK;
    if (wave == 0) {
      if (c + 1 < NCH) {   // prefetch next chunk into the other buffer
        const int j1 = j0 + CHUNK;
        tdm_load(qbL[(c + 1) & 1], (unsigned long long)(uintptr_t)(qbg + j1 * DPAD),
                 CHUNK * DPAD * 2 / 8, 0u, CHUNK * DPAD * 2 / 8, 0u, 0ull);
        tdm_load(qtL[(c + 1) & 1], (unsigned long long)(uintptr_t)(qtg + j1),
                 NQ * 2 / 8, DPAD, CHUNK * 2 / 8, DPAD, NQ * 2 / 8);
        __builtin_amdgcn_s_wait_tensorcnt(2);  // in-order: chunk c complete
      } else {
        __builtin_amdgcn_s_wait_tensorcnt(0);
      }
    }
    __syncthreads();   // LDS buffer (c&1) ready for all waves

    const _Float16* qb = &Qb[c & 1][0];
    const _Float16* qt = &Qt[c & 1][0];

    float q2[4];
    #pragma unroll
    for (int ct = 0; ct < 4; ++ct) q2[ct] = ws[WS_QW2 + j0 + ct * 16 + lt];

    // S tile 16x64 = A(16x128) x Q(128x64): 4 column tiles x 4 K-steps
    v8f S[4];
    #pragma unroll
    for (int ct = 0; ct < 4; ++ct) S[ct] = zero;
    #pragma unroll
    for (int kk = 0; kk < 4; ++kk) {
      #pragma unroll
      for (int ct = 0; ct < 4; ++ct) {
        v16h b = *(const v16h*)&qb[(ct * 16 + lt) * DPAD + kk * 32 + lh * 16];
        S[ct] = __builtin_amdgcn_wmma_f32_16x16x32_f16(false, afrag[kk], false, b,
                                                       (short)0, S[ct], false, false);
      }
    }

    // bias + online softmax update (row = g + 8*lh, col = ct*16 + lt)
    #pragma unroll
    for (int g = 0; g < 8; ++g) {
      float sv[4];
      #pragma unroll
      for (int ct = 0; ct < 4; ++ct) sv[ct] = S[ct][g] + cw1v[g] + q2[ct];
      float t = fmaxf(fmaxf(sv[0], sv[1]), fmaxf(sv[2], sv[3]));
      t = fmaxf(t, __shfl_xor(t, 1, 16));
      t = fmaxf(t, __shfl_xor(t, 2, 16));
      t = fmaxf(t, __shfl_xor(t, 4, 16));
      t = fmaxf(t, __shfl_xor(t, 8, 16));
      float mnew  = fmaxf(mrun[g], t);
      float alpha = __expf(mrun[g] - mnew);
      float p[4];
      #pragma unroll
      for (int ct = 0; ct < 4; ++ct) p[ct] = __expf(sv[ct] - mnew);
      float r = (p[0] + p[1]) + (p[2] + p[3]);
      r += __shfl_xor(r, 1, 16);
      r += __shfl_xor(r, 2, 16);
      r += __shfl_xor(r, 4, 16);
      r += __shfl_xor(r, 8, 16);
      lrun[g] = lrun[g] * alpha + r;
      mrun[g] = mnew;
      #pragma unroll
      for (int t2 = 0; t2 < NTILES; ++t2) U[t2][g] *= alpha;
      int prow = g + 8 * lh;
      #pragma unroll
      for (int ct = 0; ct < 4; ++ct)
        Plds[wave][prow * CHUNK + ct * 16 + lt] = (_Float16)p[ct];  // C->A relayout
    }
    asm volatile("" ::: "memory");  // cross-lane LDS dep within wave (DS is in-order)

    // U += P(16x64) @ Qchunk(64x112): 2 K-steps x 7 d-tiles
    #pragma unroll
    for (int kk2 = 0; kk2 < 2; ++kk2) {
      v16h pf;
      const _Float16* pb = &Plds[wave][lt * CHUNK + kk2 * 32 + lh * 16];
      v8h lo = *(const v8h*)pb;
      v8h hi = *(const v8h*)(pb + 8);
      #pragma unroll
      for (int e = 0; e < 8; ++e) { pf[e] = lo[e]; pf[e + 8] = hi[e]; }
      #pragma unroll
      for (int nt = 0; nt < NTILES; ++nt) {
        v16h bq = *(const v16h*)&qt[(nt * 16 + lt) * CHUNK + kk2 * 32 + lh * 16];
        U[nt] = __builtin_amdgcn_wmma_f32_16x16x32_f16(false, pf, false, bq,
                                                       (short)0, U[nt], false, false);
      }
    }
    __syncthreads();   // all waves done with buffer (c&1) before it is re-filled
  }

  // write row maxima m_i (uniform across each lane-half)
  #pragma unroll
  for (int g = 0; g < 8; ++g)
    if (lt == g) ws[WS_M + rowBase + lh * 8 + g] = mrun[g];

  // write U_A normalized by row sums
  #pragma unroll
  for (int nt = 0; nt < NTILES; ++nt) {
    int d = nt * 16 + lt;
    if (d < D) {
      #pragma unroll
      for (int g = 0; g < 8; ++g)
        ws[WS_U + (rowBase + lh * 8 + g) * D + d] = U[nt][g] / lrun[g];
    }
  }
}

// ---------------- kernel 2a: global softmax stats over m --------------------
__global__ void k_softmax_m(float* __restrict__ ws) {
  __shared__ float red[1024];
  int t = threadIdx.x;
  float mx = -3.0e38f;
  for (int i = t; i < NC; i += 1024) mx = fmaxf(mx, ws[WS_M + i]);
  red[t] = mx; __syncthreads();
  for (int s = 512; s > 0; s >>= 1) { if (t < s) red[t] = fmaxf(red[t], red[t + s]); __syncthreads(); }
  float M = red[0]; __syncthreads();
  float z = 0.f;
  for (int i = t; i < NC; i += 1024) z += __expf(ws[WS_M + i] - M);
  red[t] = z; __syncthreads();
  for (int s = 512; s > 0; s >>= 1) { if (t < s) red[t] += red[t + s]; __syncthreads(); }
  if (t == 0) { ws[WS_MZ] = M; ws[WS_MZ + 1] = red[0]; }
}

// ---------------- kernel 2b: h[d] = sum_i b_i * context[i][d] (deterministic)
__global__ void k_h(const float* __restrict__ ctx, float* __restrict__ ws) {
  __shared__ float red[256];
  int d = blockIdx.x, t = threadIdx.x;
  float M = ws[WS_MZ], Zinv = 1.f / ws[WS_MZ + 1];
  float acc = 0.f;
  for (int i = t; i < NC; i += 256) acc += __expf(ws[WS_M + i] - M) * ctx[i * D + d];
  red[t] = acc; __syncthreads();
  for (int s = 128; s > 0; s >>= 1) { if (t < s) red[t] += red[t + s]; __syncthreads(); }
  if (t == 0) ws[WS_H + d] = red[0] * Zinv;
}

// ---------------- kernel 3: assemble G = [C, U, C*U, C*h] -------------------
__global__ void k_out(const float* __restrict__ ctx, const float* __restrict__ ws,
                      float* __restrict__ out) {
  int i = blockIdx.x, d = threadIdx.x;
  if (d < D) {
    float c  = ctx[i * D + d];
    float u  = ws[WS_U + i * D + d];
    float hd = ws[WS_H + d];
    float* o = out + i * 4 * D;
    o[d]         = c;
    o[D + d]     = u;
    o[2 * D + d] = c * u;
    o[3 * D + d] = c * hd;
  }
}

extern "C" void kernel_launch(void* const* d_in, const int* in_sizes, int n_in,
                              void* d_out, int out_size, void* d_ws, size_t ws_size,
                              hipStream_t stream) {
  const float* ctx  = (const float*)d_in[0];
  const float* qst  = (const float*)d_in[1];
  const float* kern = (const float*)d_in[2];
  float* ws  = (float*)d_ws;
  float* out = (float*)d_out;

  hipLaunchKernelGGL(k_bias, dim3((NC + NQ + 255) / 256), dim3(256), 0, stream,
                     ctx, qst, kern, ws);
  hipLaunchKernelGGL(k_prep, dim3((NQ * DPAD + 255) / 256), dim3(256), 0, stream,
                     qst, ws);
  hipLaunchKernelGGL(k_attn, dim3(NC / RPB), dim3(256), 0, stream,
                     ctx, kern, ws);
  hipLaunchKernelGGL(k_softmax_m, dim3(1), dim3(1024), 0, stream, ws);
  hipLaunchKernelGGL(k_h, dim3(D), dim3(256), 0, stream, ctx, ws);
  hipLaunchKernelGGL(k_out, dim3(NC), dim3(128), 0, stream, ctx, ws, out);
}
